// MultiHeadAttention_19181323944259
// MI455X (gfx1250) — compile-verified, matrix operand order checked by reading, not run on hardware
//
#include <hip/hip_runtime.h>
#include <hip/hip_bf16.h>

typedef _Float16 f16;
typedef __attribute__((ext_vector_type(16))) _Float16 v16h;
typedef __attribute__((ext_vector_type(8)))  _Float16 v8h;
typedef __attribute__((ext_vector_type(8)))  float    v8f;
typedef __attribute__((ext_vector_type(4)))  float    v4f;
typedef __attribute__((ext_vector_type(4)))  int      v4i;

#define NB   4
#define NH   8
#define SEQ  2048
#define DM   256
#define DK   32
#define BS   (NB*SEQ)                 // 8192 rows
#define QK_SCALE 0.1767766952966369f // 1/sqrt(32)

// workspace byte offsets (each big buffer is exactly 4 MB)
#define OFF_XH  ((size_t)0)
#define OFF_QH  ((size_t)(4u<<20))
#define OFF_KH  ((size_t)(8u<<20))
#define OFF_VT  ((size_t)(12u<<20))
#define OFF_AT  ((size_t)(16u<<20))
#define OFF_WQ  ((size_t)(20u<<20))
#define OFF_WK  (OFF_WQ + 131072)
#define OFF_WV  (OFF_WK + 131072)
#define OFF_WO  (OFF_WV + 131072)
#define OFF_W   (OFF_WO + 131072)

// ---- CDNA5 async global->LDS path (hedged on toolchain builtin presence) ----
#if defined(__has_builtin)
# if __has_builtin(__builtin_amdgcn_global_load_async_to_lds_b128)
#  define HAVE_ASYNC_LDS 1
# endif
#endif
#ifndef HAVE_ASYNC_LDS
# define HAVE_ASYNC_LDS 0
#endif

typedef __attribute__((address_space(1))) v4i* gv4i_p;   // global v4i*
typedef __attribute__((address_space(3))) v4i* lv4i_p;   // LDS v4i*

static __device__ __forceinline__ void async_copy16(const f16* g, f16* l) {
#if HAVE_ASYNC_LDS
  __builtin_amdgcn_global_load_async_to_lds_b128(
      (gv4i_p)(v4i*)g, (lv4i_p)(v4i*)l, 0, 0);
#else
  *(v8h*)l = *(const v8h*)g;          // fallback: load + ds_store staging
#endif
}

#if HAVE_ASYNC_LDS
# if __has_builtin(__builtin_amdgcn_s_wait_asynccnt)
#  define WAIT_ASYNC(n) __builtin_amdgcn_s_wait_asynccnt(n)
# else
#  define WAIT_ASYNC(n) asm volatile("s_wait_asynccnt %0" :: "n"(n) : "memory")
# endif
#else
# define WAIT_ASYNC(n) ((void)0)
#endif

static __device__ __forceinline__ v8f wmma_f16(v16h a, v16h b, v8f c) {
  // D = A(16x32 f16) * B(32x16 f16) + C(16x16 f32)
  return __builtin_amdgcn_wmma_f32_16x16x32_f16(false, a, false, b, (short)0, c,
                                                false, false);
}

// Load one 16-bit A operand (16x32, row striped across VGPRs within a lane).
// rowptr points at this lane's 32-f16 row; hi = lane>=16.
// VGPR0-3: K=[0..7] (lo half) / [8..15] (hi half); VGPR4-7: K=[16..23]/[24..31].
static __device__ __forceinline__ v16h load_a16(const f16* rowptr, int hi) {
  v8h c0 = *(const v8h*)(rowptr + 8 * hi);
  v8h c1 = *(const v8h*)(rowptr + 16 + 8 * hi);
  return __builtin_shufflevector(c0, c1, 0,1,2,3,4,5,6,7,8,9,10,11,12,13,14,15);
}

// ---------------------------------------------------------------------------
// prep: f32->f16 conversion of x and weights, and softmax weights w[s]
// ---------------------------------------------------------------------------
__global__ void __launch_bounds__(256)
prep_kernel(const float* __restrict__ x,  const float* __restrict__ cx,
            const float* __restrict__ wq, const float* __restrict__ wk,
            const float* __restrict__ wv, const float* __restrict__ wo,
            f16* __restrict__ xh, f16* __restrict__ wqh, f16* __restrict__ wkh,
            f16* __restrict__ wvh, f16* __restrict__ woh, float* __restrict__ wbuf) {
  int i = blockIdx.x * blockDim.x + threadIdx.x;
  if (i < BS * DM) xh[i] = (f16)x[i];
  if (i < DM * DM) {
    wqh[i] = (f16)wq[i]; wkh[i] = (f16)wk[i];
    wvh[i] = (f16)wv[i]; woh[i] = (f16)wo[i];
  }
  if (i < SEQ) wbuf[i] = (i == 0) ? 0.0f : fabsf(cx[i] - cx[i - 1]);
}

// ---------------------------------------------------------------------------
// qkv: Q = (x Wq^T + bq)/sqrt(dk) row-major per head; K row-major; V transposed
// One wave computes one 16x16 output tile; 8 K-step WMMA chain.
// ---------------------------------------------------------------------------
__global__ void __launch_bounds__(256)
qkv_kernel(const f16* __restrict__ xh, const f16* __restrict__ wqh,
           const f16* __restrict__ wkh, const f16* __restrict__ wvh,
           const float* __restrict__ bq, const float* __restrict__ bk,
           const float* __restrict__ bv,
           f16* __restrict__ qh, f16* __restrict__ kh, f16* __restrict__ vt) {
  __shared__ f16 smem[8 * 16 * 16];           // 512 B per wave
  const int lane = threadIdx.x & 31, wv_ = threadIdx.x >> 5;
  const int wid = blockIdx.x * 8 + wv_;       // 24576 waves total
  const int ct = wid & 15;                    // output-col tile (16 of them)
  const int rt = (wid >> 4) & 511;            // row tile (512 of them)
  const int m  = wid >> 13;                   // 0=Q 1=K 2=V
  const int lo = lane & 15, hi = lane >> 4;

  const f16*  W    = (m == 0) ? wqh : ((m == 1) ? wkh : wvh);
  const float* bias = (m == 0) ? bq : ((m == 1) ? bk : bv);
  const f16* arow = xh + (size_t)(rt * 16 + lo) * DM;   // x rows   -> A
  const f16* brow = W  + (size_t)(ct * 16 + lo) * DM;   // W rows   -> B = W^T

  v8f acc = {};
#pragma unroll
  for (int k0 = 0; k0 < DM; k0 += 32)
    acc = wmma_f16(load_a16(arow + k0, hi), load_a16(brow + k0, hi), acc);

  const float bb = bias[ct * 16 + lo];
#pragma unroll
  for (int r = 0; r < 8; r++) acc[r] += bb;
  if (m == 0) {
#pragma unroll
    for (int r = 0; r < 8; r++) acc[r] *= QK_SCALE;
  }

  const int bidx = rt >> 7;                 // batch
  const int s0   = (rt & 127) * 16;         // seq base within batch
  const int h    = ct >> 1;                 // head
  const int d0   = (ct & 1) * 16;           // d offset within head

  if (m == 2) {
    // V^T: lane owns fixed d, consecutive s -> one packed b128 store
    v8h p;
#pragma unroll
    for (int r = 0; r < 8; r++) p[r] = (f16)acc[r];
    const int d = d0 + lo;
    *(v8h*)(vt + ((size_t)((bidx * NH + h) * DK + d)) * SEQ + s0 + 8 * hi) = p;
  } else {
    // Q/K row-major: transpose 16x16 tile through per-wave LDS region
    f16* t = smem + wv_ * 256;
#pragma unroll
    for (int r = 0; r < 8; r++) t[(r + 8 * hi) * 16 + lo] = (f16)acc[r];
    asm volatile("s_wait_dscnt 0x0" ::: "memory");
    const int row = lane >> 1, half = lane & 1;
    v8h pk = *(v8h*)(t + row * 16 + half * 8);
    f16* q = (m == 0) ? qh : kh;
    *(v8h*)(q + ((size_t)((bidx * NH + h) * SEQ + s0 + row)) * DK
              + d0 + half * 8) = pk;
  }
}

// ---------------------------------------------------------------------------
// attn: flash-style weighted softmax attention, block-cooperative K/V staging.
// Block = one (batch, head) + 8 query tiles. Each 32-key chunk of K and V^T is
// async-copied global->LDS once (256 threads x b128, double buffered, ASYNCcnt
// pipelined) and consumed by all 8 waves. scores^T = K*Q^T keeps each query's
// keys in a lane pair; P^T converts in-lane into the B operand of
// out^T = V^T*P^T. 4 WMMAs per chunk.
// ---------------------------------------------------------------------------
__global__ void __launch_bounds__(256)
attn_kernel(const f16* __restrict__ qh, const f16* __restrict__ kh,
            const f16* __restrict__ vt, const float* __restrict__ wbuf,
            f16* __restrict__ attn) {
  __shared__ f16 kbuf[2][32 * 32];            // 2 x 2 KB : K chunk rows
  __shared__ f16 vbuf[2][32 * 32];            // 2 x 2 KB : V^T chunk rows
  __shared__ f16 obuf[8 * 512];               // per-wave output staging

  const int tid = threadIdx.x;
  const int lane = tid & 31, wv_ = tid >> 5;
  const int bid = blockIdx.x;                 // 512 blocks
  const int qt8 = bid & 15, h = (bid >> 4) & 7, bidx = bid >> 7;
  const int qt = qt8 * 8 + wv_;               // this wave's query tile
  const int lo = lane & 15, hi = lane >> 4;

  const size_t hb = (size_t)(bidx * NH + h);
  const f16* qbase = qh + hb * SEQ * DK;
  const f16* kbase = kh + hb * SEQ * DK;
  const f16* vbase = vt + hb * DK * SEQ;

  // per-thread async-copy slot: one b128 (16B) per chunk.
  // threads 0-127: K rows (row=kk+crow, 64B row = 4 slots)
  // threads 128-255: V^T rows (row=d=crow, contiguous keys)
  const int crow  = (tid & 127) >> 2;         // 0..31
  const int cpart = (tid & 3) * 8;            // f16 offset within row
  const bool isK  = tid < 128;
  const f16* gsrc0 = (isK ? kbase + (size_t)crow * DK
                          : vbase + (size_t)crow * SEQ) + cpart;
  f16* lslot0 = (isK ? kbuf[0] : vbuf[0]) + crow * 32 + cpart;
  f16* lslot1 = (isK ? kbuf[1] : vbuf[1]) + crow * 32 + cpart;

  const v16h aQ = load_a16(qbase + (size_t)(qt * 16 + lo) * DK, hi); // also B=Q^T

  v8f acc0 = {}, acc1 = {};
  float m_run = -3.0e38f, l_run = 0.0f;
  const v8f zero = {};

  async_copy16(gsrc0, lslot0);                // prologue: chunk 0 -> buf 0

  for (int c = 0; c < SEQ / 32; ++c) {
    const int kk = c * 32;
    if (c + 1 < SEQ / 32) {
      const f16* gsrc = gsrc0 + (isK ? (size_t)(kk + 32) * DK : (size_t)(kk + 32));
      async_copy16(gsrc, ((c + 1) & 1) ? lslot1 : lslot0);
      WAIT_ASYNC(1);                          // chunk c landed; c+1 in flight
    } else {
      WAIT_ASYNC(0);
    }
    __syncthreads();                          // chunk c visible to all waves

    const f16* kc = kbuf[c & 1];
    const f16* vc = vbuf[c & 1];
    v16h aK0 = load_a16(kc + lo * 32, hi);          // keys kk+0..15
    v16h aK1 = load_a16(kc + (16 + lo) * 32, hi);   // keys kk+16..31
    v8f t0 = wmma_f16(aK0, aQ, zero);               // scores^T
    v8f t1 = wmma_f16(aK1, aQ, zero);

    // per-lane softmax weights: key index = kk + r + 8*hi (+16 for t1)
    v4f w0a = *(const v4f*)(wbuf + kk + 8 * hi);
    v4f w0b = *(const v4f*)(wbuf + kk + 8 * hi + 4);
    v4f w1a = *(const v4f*)(wbuf + kk + 16 + 8 * hi);
    v4f w1b = *(const v4f*)(wbuf + kk + 16 + 8 * hi + 4);

    float cm = -3.0e38f;
#pragma unroll
    for (int r = 0; r < 8; r++) cm = fmaxf(cm, fmaxf(t0[r], t1[r]));
    cm = fmaxf(cm, __shfl_xor(cm, 16, 32));
    const float mn = fmaxf(m_run, cm);
    const float alpha = __expf(m_run - mn);
    l_run *= alpha;
#pragma unroll
    for (int r = 0; r < 8; r++) { acc0[r] *= alpha; acc1[r] *= alpha; }

    float rs = 0.0f;
    v16h bP;
#pragma unroll
    for (int r = 0; r < 8; r++) {
      const float wr0 = (r < 4) ? w0a[r] : w0b[r - 4];
      const float wr1 = (r < 4) ? w1a[r] : w1b[r - 4];
      const float p0 = wr0 * __expf(t0[r] - mn);
      const float p1 = wr1 * __expf(t1[r] - mn);
      rs += p0 + p1;
      bP[r] = (f16)p0;        // B operand keys 0..15 of chunk
      bP[8 + r] = (f16)p1;    // keys 16..31
    }
    rs += __shfl_xor(rs, 16, 32);
    l_run += rs;

    v16h aV0 = load_a16(vc + lo * 32, hi);          // dims 0..15
    v16h aV1 = load_a16(vc + (16 + lo) * 32, hi);   // dims 16..31
    acc0 = wmma_f16(aV0, bP, acc0);                 // out^T
    acc1 = wmma_f16(aV1, bP, acc1);
    m_run = mn;
    __syncthreads();          // all reads of this buffer done before reuse
  }

  const float inv = 1.0f / l_run;
  f16* t = obuf + wv_ * 512;            // [q(16)][d(32)] f16 tile
  v8h h0, h1;
#pragma unroll
  for (int r = 0; r < 8; r++) {
    h0[r] = (f16)(acc0[r] * inv);
    h1[r] = (f16)(acc1[r] * inv);
  }
  *(v8h*)(t + lo * 32 + 8 * hi)      = h0;     // d = 8*hi .. +7
  *(v8h*)(t + lo * 32 + 16 + 8 * hi) = h1;     // d = 16+8*hi .. +7
  asm volatile("s_wait_dscnt 0x0" ::: "memory");
  v8h r0 = *(v8h*)(t + lo * 32 + hi * 16);
  v8h r1 = *(v8h*)(t + lo * 32 + hi * 16 + 8);
  f16* dst = attn + (size_t)(bidx * SEQ + qt * 16 + lo) * DM + h * 32 + hi * 16;
  *(v8h*)dst = r0;
  *(v8h*)(dst + 8) = r1;
}

// ---------------------------------------------------------------------------
// proj: out = attn · Wo^T + bo, f32 output, LDS-staged coalesced stores
// ---------------------------------------------------------------------------
__global__ void __launch_bounds__(256)
proj_kernel(const f16* __restrict__ attn, const f16* __restrict__ woh,
            const float* __restrict__ bo, float* __restrict__ out) {
  __shared__ float smem[8 * 256];             // 1 KB per wave
  const int lane = threadIdx.x & 31, wv_ = threadIdx.x >> 5;
  const int wid = blockIdx.x * 8 + wv_;       // 8192 waves total
  const int ct = wid & 15, rt = wid >> 4;
  const int lo = lane & 15, hi = lane >> 4;

  const f16* arow = attn + (size_t)(rt * 16 + lo) * DM;
  const f16* brow = woh + (size_t)(ct * 16 + lo) * DM;
  v8f acc = {};
#pragma unroll
  for (int k0 = 0; k0 < DM; k0 += 32)
    acc = wmma_f16(load_a16(arow + k0, hi), load_a16(brow + k0, hi), acc);

  const float bb = bo[ct * 16 + lo];
  float* t = smem + wv_ * 256;                // [row(16)][col(16)] f32
#pragma unroll
  for (int r = 0; r < 8; r++) t[(r + 8 * hi) * 16 + lo] = acc[r] + bb;
  asm volatile("s_wait_dscnt 0x0" ::: "memory");
  v4f o0 = *(v4f*)(t + lo * 16 + hi * 8);
  v4f o1 = *(v4f*)(t + lo * 16 + hi * 8 + 4);
  float* dst = out + (size_t)(rt * 16 + lo) * DM + ct * 16 + hi * 8;
  *(v4f*)dst = o0;
  *(v4f*)(dst + 4) = o1;
}

// ---------------------------------------------------------------------------
extern "C" void kernel_launch(void* const* d_in, const int* in_sizes, int n_in,
                              void* d_out, int out_size, void* d_ws, size_t ws_size,
                              hipStream_t stream) {
  const float* x  = (const float*)d_in[0];
  const float* cx = (const float*)d_in[1];
  const float* Wq = (const float*)d_in[2];
  const float* bq = (const float*)d_in[3];
  const float* Wk = (const float*)d_in[4];
  const float* bk = (const float*)d_in[5];
  const float* Wv = (const float*)d_in[6];
  const float* bv = (const float*)d_in[7];
  const float* Wo = (const float*)d_in[8];
  const float* bo = (const float*)d_in[9];

  char* ws = (char*)d_ws;
  f16*   xh   = (f16*)(ws + OFF_XH);
  f16*   qh   = (f16*)(ws + OFF_QH);
  f16*   kh   = (f16*)(ws + OFF_KH);
  f16*   vt   = (f16*)(ws + OFF_VT);
  f16*   at   = (f16*)(ws + OFF_AT);
  f16*   wqh  = (f16*)(ws + OFF_WQ);
  f16*   wkh  = (f16*)(ws + OFF_WK);
  f16*   wvh  = (f16*)(ws + OFF_WV);
  f16*   woh  = (f16*)(ws + OFF_WO);
  float* wbuf = (float*)(ws + OFF_W);

  prep_kernel<<<(BS * DM) / 256, 256, 0, stream>>>(x, cx, Wq, Wk, Wv, Wo,
                                                   xh, wqh, wkh, wvh, woh, wbuf);
  qkv_kernel<<<3072, 256, 0, stream>>>(xh, wqh, wkh, wvh, bq, bk, bv, qh, kh, vt);
  attn_kernel<<<512, 256, 0, stream>>>(qh, kh, vt, wbuf, at);
  proj_kernel<<<1024, 256, 0, stream>>>(at, woh, bo, (float*)d_out);
}
